// BiMlp_adapscaling_1w32a_1305670058376
// MI455X (gfx1250) — matrix operand order repaired and checked by reading, not compiled, not run on hardware
//
#include <hip/hip_runtime.h>
#include <hip/hip_bf16.h>

// ---------------------------------------------------------------------------
// BiMLP (binary-weight MLP) for MI455X / gfx1250, wave32 + WMMA bf16.
//   h   = gelu( alpha1 * (x @ sign(w1)^T) + b1 )   (alpha folded in epilogue: exact)
//   out =        alpha2 * (h @ sign(w2)^T) + b2
// Shapes: M=12544 (=98*128), D=1024, H=4096. All tile dims divide evenly.
// GEMM: macro tile 128(M) x 256(N), K-step 32, 8 waves of 64x64 (4x4 WMMA tiles),
// double-buffered LDS fed by GLOBAL_LOAD_ASYNC_TO_LDS_B128 when available.
// ---------------------------------------------------------------------------

typedef __attribute__((ext_vector_type(16))) __bf16 v16bf;
typedef __attribute__((ext_vector_type(8)))  float  v8f;
typedef int v4i __attribute__((vector_size(16)));   // matches builtin param type

#define AS1 __attribute__((address_space(1)))
#define AS3 __attribute__((address_space(3)))

#if __has_builtin(__builtin_amdgcn_global_load_async_to_lds_b128) && \
    __has_builtin(__builtin_amdgcn_s_wait_asynccnt)
#define USE_ASYNC 1
#else
#define USE_ASYNC 0
#endif

struct B32x2 { uint4 lo, hi; };   // 32 bytes -> bit_cast to v16bf

__device__ __forceinline__ unsigned short f2bf_rne(float f) {
    unsigned u = __builtin_bit_cast(unsigned, f);
    unsigned r = u + 0x7FFFu + ((u >> 16) & 1u);   // round-to-nearest-even
    return (unsigned short)(r >> 16);
}

// Branchless GELU (tanh form; error ~1e-3, below bf16 storage granularity).
// tanh(u) = 1 - 2/(1+e^{2u}) -> v_exp_f32 + rcp, no divergence.
__device__ __forceinline__ float gelu_f(float v) {
    float u = 0.7978845608028654f * (v + 0.044715f * v * v * v);
    float e = __expf(2.0f * u);
    float t = 1.0f - 2.0f / (e + 1.0f);
    return 0.5f * v * (1.0f + t);
}

#if USE_ASYNC
__device__ __forceinline__ void async_cp16(const unsigned short* g, unsigned short* l) {
    __builtin_amdgcn_global_load_async_to_lds_b128((AS1 v4i*)g, (AS3 v4i*)l, 0, 0);
}
#endif

// ---------------------------------------------------------------------------
// Per-output-channel binarization: alpha[row] = mean(|w[row,:]|),
// s[row,k] = sign(w[row,k]) as bf16 bits (+1 / -1 / 0). One block per row.
// ---------------------------------------------------------------------------
__global__ __launch_bounds__(256) void binarize_rows(
    const float* __restrict__ w, unsigned short* __restrict__ s,
    float* __restrict__ alpha, int K)
{
    const int row = blockIdx.x;
    const float* wr = w + (size_t)row * K;
    unsigned short* sr = s + (size_t)row * K;
    float acc = 0.f;
    for (int k = threadIdx.x; k < K; k += 256) {
        float v = wr[k];
        acc += fabsf(v);
        sr[k] = (v > 0.f) ? (unsigned short)0x3F80u
                          : ((v < 0.f) ? (unsigned short)0xBF80u : (unsigned short)0u);
    }
    __shared__ float red[256];
    red[threadIdx.x] = acc;
    __syncthreads();
    #pragma unroll
    for (int off = 128; off > 0; off >>= 1) {
        if (threadIdx.x < off) red[threadIdx.x] += red[threadIdx.x + off];
        __syncthreads();
    }
    if (threadIdx.x == 0) alpha[row] = red[0] / (float)K;
}

// ---------------------------------------------------------------------------
// f32 -> bf16 (RNE), 4 elements/thread. n divisible by 4.
// ---------------------------------------------------------------------------
__global__ __launch_bounds__(256) void f32_to_bf16_vec(
    const float* __restrict__ in, unsigned short* __restrict__ out, long n)
{
    long i = ((long)blockIdx.x * 256 + threadIdx.x) * 4;
    if (i >= n) return;
    float4 v = *(const float4*)(in + i);
    unsigned lo = (unsigned)f2bf_rne(v.x) | ((unsigned)f2bf_rne(v.y) << 16);
    unsigned hi = (unsigned)f2bf_rne(v.z) | ((unsigned)f2bf_rne(v.w) << 16);
    uint2 p; p.x = lo; p.y = hi;
    *(uint2*)(out + i) = p;
}

// ---------------------------------------------------------------------------
// Tiled bf16 WMMA GEMM: acc[m,n] = sum_k A[m,k]*B[n,k], epilogue
// val = alpha[n]*acc + bias[n]; GELU_BF16_OUT ? gelu->bf16 : f32.
// Block: 256 threads (8 waves), tile 128x256, wave (wm=wv&1, wn=wv>>1)
// owns 64(M)x64(N) = 4x4 tiles of v_wmma_f32_16x16x32_bf16.
// ---------------------------------------------------------------------------
template <int GELU_BF16_OUT>
__global__ __launch_bounds__(256) void gemm_bin(
    const unsigned short* __restrict__ A,   // [M,K] bf16 activations
    const unsigned short* __restrict__ Bm,  // [N,K] bf16 sign weights
    const float* __restrict__ alpha,        // [N]
    const float* __restrict__ bias,         // [N]
    void* __restrict__ Out,                 // [M,N] bf16 or f32
    int M, int N, int K)
{
    // [2 buffers][rows][32 + 8 pad bf16]: 80B row stride -> conflict-free frag reads
    __shared__ __align__(16) unsigned short As[2][128 * 40];
    __shared__ __align__(16) unsigned short Bs[2][256 * 40];

    const int tid  = threadIdx.x;
    const int lane = tid & 31;
    const int wv   = tid >> 5;
    const int wm   = wv & 1;          // which 64-row M half
    const int wn   = wv >> 1;         // which 64-col N slab (0..3)
    const int m0   = blockIdx.y * 128;
    const int n0   = blockIdx.x * 256;

    // ---- staging indices: each thread owns (row = tid>>2 (+64j), 8-bf16 chunk c8)
    const int rowT = tid >> 2;        // 0..63
    const int c8   = tid & 3;
    const unsigned short* AgBase = A  + (size_t)(m0 + rowT) * K + c8 * 8;
    const unsigned short* BgBase = Bm + (size_t)(n0 + rowT) * K + c8 * 8;
    const size_t stride64 = (size_t)64 * K;   // 64 rows, in ushorts

    // ---- fragment read indices (uint4 units), per ISA bf16 A/B VGPR layouts
    const int aoff = (lane < 16) ? 0 : 1;   // A: K base 0/8   (second half +2 uint4)
    const int boff = (lane < 16) ? 0 : 1;   // B: K base 0/8   (second half +2 uint4)
    const int aRow = wm * 64 + (lane & 15);
    const int bRow = wn * 64 + (lane & 15);

    v8f acc[4][4];
    #pragma unroll
    for (int mt = 0; mt < 4; ++mt)
        #pragma unroll
        for (int nt = 0; nt < 4; ++nt)
            #pragma unroll
            for (int r = 0; r < 8; ++r) acc[mt][nt][r] = 0.f;

    const int KT = K >> 5;

#if USE_ASYNC
    // ---------------- async-copy staging (GLOBAL_LOAD_ASYNC_TO_LDS_B128) -----
    auto stage = [&](int buf, int koff) {
        #pragma unroll
        for (int j = 0; j < 2; ++j)
            async_cp16(AgBase + (size_t)j * stride64 + koff,
                       As[buf] + (rowT + 64 * j) * 40 + c8 * 8);
        #pragma unroll
        for (int j = 0; j < 4; ++j)
            async_cp16(BgBase + (size_t)j * stride64 + koff,
                       Bs[buf] + (rowT + 64 * j) * 40 + c8 * 8);
    };
    stage(0, 0);
    __builtin_amdgcn_s_wait_asynccnt(0);
    __syncthreads();
#else
    // ---------------- fallback: register-staged copy -------------------------
    {
        uint4 ra[2], rb[4];
        #pragma unroll
        for (int j = 0; j < 2; ++j) ra[j] = *(const uint4*)(AgBase + (size_t)j * stride64);
        #pragma unroll
        for (int j = 0; j < 4; ++j) rb[j] = *(const uint4*)(BgBase + (size_t)j * stride64);
        uint4* As4w = (uint4*)As[0];
        uint4* Bs4w = (uint4*)Bs[0];
        #pragma unroll
        for (int j = 0; j < 2; ++j) As4w[(rowT + 64 * j) * 5 + c8] = ra[j];
        #pragma unroll
        for (int j = 0; j < 4; ++j) Bs4w[(rowT + 64 * j) * 5 + c8] = rb[j];
    }
    __syncthreads();
#endif

    for (int kk = 0; kk < KT; ++kk) {
        const int buf = kk & 1;
        const bool haveNext = (kk + 1 < KT);

#if USE_ASYNC
        if (haveNext) stage(buf ^ 1, (kk + 1) * 32);  // overlaps with compute below
#else
        uint4 ra[2], rb[4];
        if (haveNext) {
            const int off = (kk + 1) * 32;
            #pragma unroll
            for (int j = 0; j < 2; ++j) ra[j] = *(const uint4*)(AgBase + (size_t)j * stride64 + off);
            #pragma unroll
            for (int j = 0; j < 4; ++j) rb[j] = *(const uint4*)(BgBase + (size_t)j * stride64 + off);
        }
        if (kk + 2 < KT) {  // global_prefetch_b8 two steps ahead
            const int poff = (kk + 2) * 32;
            __builtin_prefetch(AgBase + poff, 0, 0);
            __builtin_prefetch(BgBase + poff, 0, 0);
        }
#endif

        // ---- fragments from LDS (ds_load_b128 pairs, ISA-layout exact)
        const uint4* As4 = (const uint4*)As[buf];
        const uint4* Bs4 = (const uint4*)Bs[buf];
        v16bf af[4], bfr[4];
        #pragma unroll
        for (int mt = 0; mt < 4; ++mt) {
            const int base = (aRow + mt * 16) * 5 + aoff;
            B32x2 t; t.lo = As4[base]; t.hi = As4[base + 2];
            af[mt] = __builtin_bit_cast(v16bf, t);
        }
        #pragma unroll
        for (int nt = 0; nt < 4; ++nt) {
            const int base = (bRow + nt * 16) * 5 + boff;
            B32x2 t; t.lo = Bs4[base]; t.hi = Bs4[base + 2];
            bfr[nt] = __builtin_bit_cast(v16bf, t);
        }

        // ---- 16 x v_wmma_f32_16x16x32_bf16
        #pragma unroll
        for (int mt = 0; mt < 4; ++mt)
            #pragma unroll
            for (int nt = 0; nt < 4; ++nt)
                acc[mt][nt] = __builtin_amdgcn_wmma_f32_16x16x32_bf16(
                    false, af[mt], false, bfr[nt], (short)0, acc[mt][nt], false, false);

        // ---- publish next buffer
        if (haveNext) {
#if USE_ASYNC
            __builtin_amdgcn_s_wait_asynccnt(0);
#else
            uint4* As4w = (uint4*)As[buf ^ 1];
            uint4* Bs4w = (uint4*)Bs[buf ^ 1];
            #pragma unroll
            for (int j = 0; j < 2; ++j) As4w[(rowT + 64 * j) * 5 + c8] = ra[j];
            #pragma unroll
            for (int j = 0; j < 4; ++j) Bs4w[(rowT + 64 * j) * 5 + c8] = rb[j];
#endif
            __syncthreads();
        }
    }

    // ---- epilogue: D layout = VGPR r -> (M = r + (lane>=16 ? 8:0), N = lane&15)
    const int nl    = lane & 15;
    const int mbase = m0 + wm * 64 + ((lane < 16) ? 0 : 8);
    #pragma unroll
    for (int nt = 0; nt < 4; ++nt) {
        const int n  = n0 + wn * 64 + nt * 16 + nl;
        const float al = alpha[n];
        const float bi = bias[n];
        #pragma unroll
        for (int mt = 0; mt < 4; ++mt) {
            const int mrow = mbase + mt * 16;
            #pragma unroll
            for (int r = 0; r < 8; ++r) {
                float v = al * acc[mt][nt][r] + bi;
                if (GELU_BF16_OUT) {
                    ((unsigned short*)Out)[(size_t)(mrow + r) * N + n] = f2bf_rne(gelu_f(v));
                } else {
                    ((float*)Out)[(size_t)(mrow + r) * N + n] = v;
                }
            }
        }
    }
}

// ---------------------------------------------------------------------------
// Host-side orchestration
// ---------------------------------------------------------------------------
extern "C" void kernel_launch(void* const* d_in, const int* in_sizes, int n_in,
                              void* d_out, int out_size, void* d_ws, size_t ws_size,
                              hipStream_t stream) {
    (void)in_sizes; (void)n_in; (void)out_size; (void)ws_size;
    const int M = 64 * 196;   // 12544 = 98 * 128
    const int D = 1024;
    const int H = 4096;

    const float* x  = (const float*)d_in[0];
    const float* w1 = (const float*)d_in[1];
    const float* b1 = (const float*)d_in[2];
    const float* w2 = (const float*)d_in[3];
    const float* b2 = (const float*)d_in[4];

    char* ws = (char*)d_ws;
    size_t off = 0;
    auto carve = [&](size_t bytes) {
        void* p = ws + off;
        off += (bytes + 255) & ~(size_t)255;
        return p;
    };
    unsigned short* xb = (unsigned short*)carve((size_t)M * D * 2);  // x as bf16
    unsigned short* s1 = (unsigned short*)carve((size_t)H * D * 2);  // sign(w1) bf16
    unsigned short* s2 = (unsigned short*)carve((size_t)D * H * 2);  // sign(w2) bf16
    float*          a1 = (float*)carve((size_t)H * 4);               // alpha1
    float*          a2 = (float*)carve((size_t)D * 4);               // alpha2
    unsigned short* hb = (unsigned short*)carve((size_t)M * H * 2);  // hidden bf16

    binarize_rows<<<H, 256, 0, stream>>>(w1, s1, a1, D);
    binarize_rows<<<D, 256, 0, stream>>>(w2, s2, a2, H);
    {
        long n = (long)M * D;
        long blocks = (n / 4 + 255) / 256;
        f32_to_bf16_vec<<<(int)blocks, 256, 0, stream>>>(x, xb, n);
    }

    // fc1 + GELU -> hb (bf16): grid 16 x 98
    gemm_bin<1><<<dim3(H / 256, M / 128), 256, 0, stream>>>(
        xb, s1, a1, b1, (void*)hb, M, H, D);
    // fc2 -> out (f32): grid 4 x 98
    gemm_bin<0><<<dim3(D / 256, M / 128), 256, 0, stream>>>(
        hb, s2, a2, b2, d_out, M, D, H);
}
